// CM_TPM_53137335386823
// MI455X (gfx1250) — matrix operand order repaired and checked by reading, not compiled, not run on hardware
//
#include <hip/hip_runtime.h>
#include <hip/hip_bf16.h>

typedef __attribute__((ext_vector_type(16))) __bf16 v16bf;
typedef __attribute__((ext_vector_type(8)))  float  v8f;
typedef __attribute__((ext_vector_type(8)))  unsigned int v8u;

union Frag { v8u u; v16bf b; };

// --- bf16 split helpers: a = hi(trunc) + lo(rne of residual) ----------------
__device__ __forceinline__ unsigned int pack_hi2(float a0, float a1) {
  unsigned int u0 = __float_as_uint(a0) >> 16;
  unsigned int u1 = __float_as_uint(a1) >> 16;
  return u0 | (u1 << 16);
}
__device__ __forceinline__ unsigned int pack_lo2(float a0, float a1) {
  float h0 = __uint_as_float(__float_as_uint(a0) & 0xFFFF0000u);
  float h1 = __uint_as_float(__float_as_uint(a1) & 0xFFFF0000u);
  float r0 = a0 - h0, r1 = a1 - h1;
  unsigned int v0 = __float_as_uint(r0);
  unsigned int v1 = __float_as_uint(r1);
  v0 = (v0 + 0x7FFFu + ((v0 >> 16) & 1u)) >> 16;
  v1 = (v1 + 0x7FFFu + ((v1 >> 16) & 1u)) >> 16;
  return (v0 & 0xFFFFu) | (v1 << 16);
}

// CDNA5 async direct-to-LDS copy: 16 bytes per lane, no data VGPRs,
// tracked by ASYNCcnt. ldsOff = raw LDS byte address (VDST), gaddr = 64-bit VA.
__device__ __forceinline__ void asyncLoad16(unsigned int ldsOff, const void* gaddr) {
  asm volatile("global_load_async_to_lds_b128 %0, %1, off"
               :: "v"(ldsOff), "v"((unsigned long long)(uintptr_t)gaddr)
               : "memory");
}
__device__ __forceinline__ void waitAsync0() {
  asm volatile("s_wait_asynccnt 0x0" ::: "memory");
}

// ---------------------------------------------------------------------------
// Stage 1: hidden = relu(z @ W1 + b1)   (K=2048, L=128, H=64)
__global__ void CM_mlp1(const float* __restrict__ z, const float* __restrict__ W1,
                        const float* __restrict__ b1, float* __restrict__ hidden) {
  int idx = blockIdx.x * 256 + threadIdx.x;        // < 2048*64
  int k = idx >> 6, h = idx & 63;
  float s = b1[h];
  #pragma unroll 8
  for (int l = 0; l < 128; ++l) s += z[k * 128 + l] * W1[l * 64 + h];
  hidden[idx] = fmaxf(s, 0.0f);
}

// Stage 2: phi = hidden @ W2 + b2       (K=2048, H=64, D=512)
__global__ void CM_mlp2(const float* __restrict__ hidden, const float* __restrict__ W2,
                        const float* __restrict__ b2, float* __restrict__ phi) {
  int idx = blockIdx.x * 256 + threadIdx.x;        // < 2048*512
  int k = idx >> 9, d = idx & 511;
  float s = b2[d];
  #pragma unroll 8
  for (int h = 0; h < 64; ++h) s += hidden[k * 64 + h] * W2[h * 512 + d];
  phi[idx] = s;
}

// Row squared-norms: out[row] = sum(mat[row,:]^2), one block/row
__global__ void CM_rownorm(const float* __restrict__ mat, float* __restrict__ out, int D) {
  __shared__ float sred[8];
  int row = blockIdx.x, t = threadIdx.x;
  float s = 0.0f;
  for (int d = t; d < D; d += 256) { float v = mat[(size_t)row * D + d]; s += v * v; }
  #pragma unroll
  for (int off = 16; off > 0; off >>= 1) s += __shfl_down(s, off, 32);
  if ((t & 31) == 0) sred[t >> 5] = s;
  __syncthreads();
  if (t < 32) {
    float v = (t < 8) ? sred[t] : 0.0f;
    #pragma unroll
    for (int off = 4; off > 0; off >>= 1) v += __shfl_down(v, off, 32);
    if (t == 0) out[row] = v;
  }
}

// One-time fp32 -> packed bf16 hi/lo split (2 elements -> 1 dword each array).
__global__ void CM_split(const float* __restrict__ in, unsigned int* __restrict__ hi,
                         unsigned int* __restrict__ lo) {
  int i = blockIdx.x * 256 + threadIdx.x;          // over dword pairs
  float2 a = ((const float2*)in)[i];
  hi[i] = pack_hi2(a.x, a.y);
  lo[i] = pack_lo2(a.x, a.y);
}

// ---------------------------------------------------------------------------
// Fused kernel: 128x128 tile of S = phi @ x^T via split-bf16 WMMA, then
// partial = sum_{k,b in tile} w[k] * exp(-(p2[k]+x2[b]-2*S[k,b]))
//
// Staging uses GLOBAL_LOAD_ASYNC_TO_LDS_B128 (no data VGPRs, ASYNCcnt) into a
// double-buffered LDS tile: iteration it issues asyncs for tile it+1 into the
// other buffer while 24 v_wmma consume the current one; one s_wait_asynccnt
// + barrier per iteration. LDS row stride 20 dwords (80 B: b128-aligned,
// bank-spread). Single combined __shared__ block so LDS byte offsets are raw.
#define TILE_BYTES   10240          // 128 rows * 20 dwords * 4 B
#define BUF_BYTES    40960          // 4 tile arrays (Ah, Al, Bh, Bl)
#define P2_OFF       81920          // after 2 buffers
#define W_OFF        82432
#define WSUM_OFF     82944
#define SMEM_BYTES   83008

__global__ __launch_bounds__(256, 1)
void CM_pc_gemm(const unsigned int* __restrict__ phiH, const unsigned int* __restrict__ phiL,
                const unsigned int* __restrict__ xH,   const unsigned int* __restrict__ xL,
                const float* __restrict__ p2,  const float* __restrict__ x2,
                const float* __restrict__ w,   float* __restrict__ partials) {
  constexpr int DU = 256;                  // dwords per row (D/2)
  constexpr int NIT = 16;                  // 512 / 32
  __shared__ __align__(16) unsigned char smem[SMEM_BYTES];

  const int t    = threadIdx.x;
  const int lane = t & 31;
  const int wave = t >> 5;
  const int half = lane >> 4;              // lanes 0-15 vs 16-31
  const int l16  = lane & 15;

  const int kTile = blockIdx.y * 128;
  const int bTile = blockIdx.x * 128;
  const int wr = wave & 3;                 // 32-row strip
  const int wc = wave >> 2;                // 64-col strip

  v8f acc[2][4];
  #pragma unroll
  for (int i = 0; i < 2; ++i)
    #pragma unroll
    for (int j = 0; j < 4; ++j)
      #pragma unroll
      for (int e = 0; e < 8; ++e) acc[i][j][e] = 0.0f;

  // Staging assignment: thread t covers row ldRow, dword-halfcolumn ldColU.
  const int ldRow  = t >> 1;               // 0..127
  const int ldColU = (t & 1) * 8;          // 0 / 8 dwords (16 bf16)

  const unsigned int* gAh = phiH + (size_t)(kTile + ldRow) * DU + ldColU;
  const unsigned int* gAl = phiL + (size_t)(kTile + ldRow) * DU + ldColU;
  const unsigned int* gBh = xH   + (size_t)(bTile + ldRow) * DU + ldColU;
  const unsigned int* gBl = xL   + (size_t)(bTile + ldRow) * DU + ldColU;

  const unsigned int sOffB = (unsigned)(ldRow * 80 + ldColU * 4);  // LDS bytes

  // issue one tile's 8 async b128 copies (4 arrays x 2 chunks)
  auto stage = [&](int tile, int buf) {
    const int go = tile * 16;              // 16 dwords per 32-bf16 chunk
    const unsigned int base = (unsigned)buf * BUF_BYTES + sOffB;
    #pragma unroll
    for (int q = 0; q < 2; ++q) {
      asyncLoad16(base + 0 * TILE_BYTES + q * 16, gAh + go + 4 * q);
      asyncLoad16(base + 1 * TILE_BYTES + q * 16, gAl + go + 4 * q);
      asyncLoad16(base + 2 * TILE_BYTES + q * 16, gBh + go + 4 * q);
      asyncLoad16(base + 3 * TILE_BYTES + q * 16, gBl + go + 4 * q);
    }
  };

  stage(0, 0);
  waitAsync0();
  __syncthreads();

  #pragma unroll 1
  for (int it = 0; it < NIT; ++it) {
    const int cur = it & 1;
    if (it + 1 < NIT) stage(it + 1, 1 - cur);   // overlaps with WMMA below

    const unsigned int* bufU = (const unsigned int*)(smem + cur * BUF_BYTES);
    const unsigned int* sAh = bufU;
    const unsigned int* sAl = bufU + 2560;      // TILE_BYTES/4
    const unsigned int* sBh = bufU + 5120;
    const unsigned int* sBl = bufU + 7680;

    // A fragments for both 16-row subtiles, 16x32 bf16 layout (ISA 7.12.2):
    //   lanes 0-15: v0-3 K=0..7, v4-7 K=16..23 ; lanes 16-31: +8
    Frag ah[2], al[2];
    #pragma unroll
    for (int ti = 0; ti < 2; ++ti) {
      const int m = wr * 32 + ti * 16 + l16;
      const unsigned int* rah = &sAh[m * 20];
      const unsigned int* ral = &sAl[m * 20];
      #pragma unroll
      for (int j = 0; j < 4; ++j) {
        ah[ti].u[j]     = rah[half * 4 + j];
        ah[ti].u[4 + j] = rah[8 + half * 4 + j];
        al[ti].u[j]     = ral[half * 4 + j];
        al[ti].u[4 + j] = ral[8 + half * 4 + j];
      }
    }
    #pragma unroll
    for (int tj = 0; tj < 4; ++tj) {
      // B fragment, 32x16: lanes 0-15 K=0..7,16..23, lanes 16-31 +8, N=lane%16
      const int n = wc * 64 + tj * 16 + l16;
      const unsigned int* rbh = &sBh[n * 20];
      const unsigned int* rbl = &sBl[n * 20];
      Frag bh, bl;
      #pragma unroll
      for (int j = 0; j < 8; ++j) { bh.u[j] = rbh[(j >> 2) * 8 + half * 4 + (j & 3)]; bl.u[j] = rbl[(j >> 2) * 8 + half * 4 + (j & 3)]; }
      #pragma unroll
      for (int ti = 0; ti < 2; ++ti) {
        acc[ti][tj] = __builtin_amdgcn_wmma_f32_16x16x32_bf16(
            false, ah[ti].b, false, bh.b, (short)0, acc[ti][tj], false, false);
        acc[ti][tj] = __builtin_amdgcn_wmma_f32_16x16x32_bf16(
            false, ah[ti].b, false, bl.b, (short)0, acc[ti][tj], false, false);
        acc[ti][tj] = __builtin_amdgcn_wmma_f32_16x16x32_bf16(
            false, al[ti].b, false, bh.b, (short)0, acc[ti][tj], false, false);
      }
    }

    waitAsync0();                          // next tile landed in other buffer
    __syncthreads();                       // all waves done reading/writing
  }

  float* sP2   = (float*)(smem + P2_OFF);
  float* sW    = (float*)(smem + W_OFF);
  float* sWsum = (float*)(smem + WSUM_OFF);
  if (t < 128) { sP2[t] = p2[kTile + t]; sW[t] = w[kTile + t]; }
  __syncthreads();

  // Epilogue: C/D layout -> lane holds col n=l16, rows r + 8*half in v0..7
  float lsum = 0.0f;
  #pragma unroll
  for (int ti = 0; ti < 2; ++ti) {
    #pragma unroll
    for (int tj = 0; tj < 4; ++tj) {
      const int nl  = wc * 64 + tj * 16 + l16;
      const float xx2 = x2[bTile + nl];
      const int mbase = wr * 32 + ti * 16 + half * 8;
      #pragma unroll
      for (int r = 0; r < 8; ++r) {
        const int ml = mbase + r;
        const float sq = sP2[ml] + xx2 - 2.0f * acc[ti][tj][r];
        lsum += sW[ml] * __expf(-sq);
      }
    }
  }
  #pragma unroll
  for (int off = 16; off > 0; off >>= 1) lsum += __shfl_down(lsum, off, 32);
  if (lane == 0) sWsum[wave] = lsum;
  __syncthreads();
  if (wave == 0) {
    float v = (lane < 8) ? sWsum[lane] : 0.0f;
    #pragma unroll
    for (int off = 4; off > 0; off >>= 1) v += __shfl_down(v, off, 32);
    if (lane == 0) partials[blockIdx.y * gridDim.x + blockIdx.x] = v;
  }
}

// Final deterministic reduction of block partials; writes mean scalar.
__global__ void CM_reduce(const float* __restrict__ partials, int n, float scale,
                          float* __restrict__ out) {
  __shared__ float sred[8];
  int t = threadIdx.x;
  float s = 0.0f;
  for (int i = t; i < n; i += 256) s += partials[i];
  #pragma unroll
  for (int off = 16; off > 0; off >>= 1) s += __shfl_down(s, off, 32);
  if ((t & 31) == 0) sred[t >> 5] = s;
  __syncthreads();
  if (t < 32) {
    float v = (t < 8) ? sred[t] : 0.0f;
    #pragma unroll
    for (int off = 4; off > 0; off >>= 1) v += __shfl_down(v, off, 32);
    if (t == 0) out[0] = v * scale;
  }
}

extern "C" void kernel_launch(void* const* d_in, const int* in_sizes, int n_in,
                              void* d_out, int out_size, void* d_ws, size_t ws_size,
                              hipStream_t stream) {
  (void)n_in; (void)out_size; (void)ws_size;
  const float* x  = (const float*)d_in[0];  // (8192,512)
  const float* z  = (const float*)d_in[1];  // (2048,128)
  const float* w  = (const float*)d_in[2];  // (2048,)
  const float* W1 = (const float*)d_in[3];  // (128,64)
  const float* b1 = (const float*)d_in[4];  // (64,)
  const float* W2 = (const float*)d_in[5];  // (64,512)
  const float* b2 = (const float*)d_in[6];  // (512,)
  float* out = (float*)d_out;

  const int D = 512;
  const int B = in_sizes[0] / D;            // 8192
  const int K = in_sizes[2];                // 2048

  // Workspace layout (all chunks 16B-aligned)
  char* ws = (char*)d_ws;
  float* hidden   = (float*)ws;                         ws += (size_t)K * 64 * 4;
  float* phi      = (float*)ws;                         ws += (size_t)K * D * 4;
  float* p2       = (float*)ws;                         ws += (size_t)K * 4;
  float* x2       = (float*)ws;                         ws += (size_t)B * 4;
  float* partials = (float*)ws;                         ws += 4096;
  unsigned int* phiH = (unsigned int*)ws;               ws += (size_t)K * D * 2;
  unsigned int* phiL = (unsigned int*)ws;               ws += (size_t)K * D * 2;
  unsigned int* xH   = (unsigned int*)ws;               ws += (size_t)B * D * 2;
  unsigned int* xL   = (unsigned int*)ws;

  CM_mlp1<<<(K * 64) / 256, 256, 0, stream>>>(z, W1, b1, hidden);
  CM_mlp2<<<(K * D) / 256, 256, 0, stream>>>(hidden, W2, b2, phi);
  CM_rownorm<<<K, 256, 0, stream>>>(phi, p2, D);
  CM_rownorm<<<B, 256, 0, stream>>>(x, x2, D);
  CM_split<<<(K * D / 2) / 256, 256, 0, stream>>>(phi, phiH, phiL);
  CM_split<<<(B * D / 2) / 256, 256, 0, stream>>>(x, xH, xL);

  dim3 grid(B / 128, K / 128);              // (64, 16)
  CM_pc_gemm<<<grid, 256, 0, stream>>>(phiH, phiL, xH, xL, p2, x2, w, partials);

  const int nPart = (B / 128) * (K / 128);  // 1024
  CM_reduce<<<1, 256, 0, stream>>>(partials, nPart, 1.0f / (float)B, out);
}
